// RSSM_64098091925666
// MI455X (gfx1250) — compile-verified, matrix operand order checked by reading, not run on hardware
//
#include <hip/hip_runtime.h>

#define AK 32
#define DK 1024
#define LK 1024
#define HK 2048
#define BB 256
#define TT 64
#define G3 3072
#define BT 16384   // B*T

typedef __attribute__((ext_vector_type(16))) __bf16       v16bf;
typedef __attribute__((ext_vector_type(8)))  float        v8f;
typedef __attribute__((ext_vector_type(4)))  unsigned int u32x4;

static __device__ __forceinline__ unsigned short f2bf(float f) {
  unsigned u = __float_as_uint(f);
  u += 0x7fffu + ((u >> 16) & 1u);   // round-to-nearest-even
  return (unsigned short)(u >> 16);
}
static __device__ __forceinline__ float bf2f(unsigned short h) {
  return __uint_as_float(((unsigned)h) << 16);
}
static __device__ __forceinline__ float sigf(float x) {
  return 1.f / (1.f + __expf(-x));
}

// ---- CDNA5 async global->LDS copy (ASYNCcnt path) ----
static __device__ __forceinline__ void async_ld_b128(unsigned lds_off, const void* gptr) {
  asm volatile("global_load_async_to_lds_b128 %0, %1, off"
               :: "v"(lds_off), "v"(gptr)
               : "memory");
}
#if __has_builtin(__builtin_amdgcn_s_wait_asynccnt)
#define WAIT_ASYNC(n) __builtin_amdgcn_s_wait_asynccnt(n)
#else
#define WAIT_ASYNC(n) asm volatile("s_wait_asynccnt %0" :: "i"(n) : "memory")
#endif

static __device__ __forceinline__ unsigned lds_off32(const void* p) {
  // addrspace(3)->generic cast keeps the LDS byte offset in the low 32 bits
  return (unsigned)(unsigned long long)p;
}

union FragU { v16bf v; u32x4 q[2]; };

// Load a 16x32 bf16 fragment from a row-major [rows x 32] LDS tile,
// following the CDNA5 16-bit A/B VGPR layout:
// lanes 0-15: row = lane,    K = {0..7, 16..23}
// lanes 16-31: row = lane-16, K = {8..15, 24..31}
static __device__ __forceinline__ v16bf frag_ld(const unsigned short* s, int row0, int lane) {
  int r  = row0 + (lane & 15);
  int kb = (lane & 16) ? 8 : 0;
  FragU f;
  f.q[0] = *(const u32x4*)(s + r * 32 + kb);
  f.q[1] = *(const u32x4*)(s + r * 32 + kb + 16);
  return f.v;
}

// C[M,N] = A[M,K] * W[N,K]^T (+bias) (+relu). A,W bf16 (ushort), C f32, Cbf bf16 copy.
// M,N multiples of 128; K multiple of 32. Double-buffered async LDS staging.
__global__ __launch_bounds__(256)
void k_gemm(const unsigned short* __restrict__ Am, const unsigned short* __restrict__ Wm,
            const float* __restrict__ bias, float* __restrict__ C,
            unsigned short* __restrict__ Cbf, int M, int N, int K, int relu) {
  __shared__ unsigned short sA[2][128 * 32];
  __shared__ unsigned short sB[2][128 * 32];
  const int tid  = threadIdx.x;
  const int lane = tid & 31;
  const int wid  = tid >> 5;
  const int wm   = (wid >> 2) * 64;   // wave row offset in block tile
  const int wn   = (wid & 3) * 32;    // wave col offset in block tile
  const int bm   = blockIdx.y * 128;
  const int bn   = blockIdx.x * 128;
  const int nk   = K >> 5;

  v8f acc[4][2];
  for (int a = 0; a < 4; ++a)
    for (int b = 0; b < 2; ++b)
      for (int i = 0; i < 8; ++i) acc[a][b][i] = 0.f;

  // Stage kt into LDS buffer `buf`: 512 16B chunks per tile, 2 chunks/thread/tile
  // => 4 async instructions per wave per stage.
  auto issue = [&](int kt, int buf) {
    int k0 = kt * 32;
    for (int c = tid; c < 512; c += 256) {
      int row = c >> 2, cc = (c & 3) * 8;
      async_ld_b128(lds_off32(&sA[buf][row * 32 + cc]),
                    Am + (size_t)(bm + row) * K + k0 + cc);
      async_ld_b128(lds_off32(&sB[buf][row * 32 + cc]),
                    Wm + (size_t)(bn + row) * K + k0 + cc);
    }
  };

  issue(0, 0);
  for (int kt = 0; kt < nk; ++kt) {
    const int buf = kt & 1;
    if (kt + 1 < nk) {
      issue(kt + 1, buf ^ 1);   // prefetch next stage into other buffer
      WAIT_ASYNC(4);            // wait for current stage (4 newest still in flight)
    } else {
      WAIT_ASYNC(0);
    }
    __syncthreads();

    v16bf bfr[2];
#pragma unroll
    for (int tn = 0; tn < 2; ++tn) bfr[tn] = frag_ld(sB[buf], wn + tn * 16, lane);
#pragma unroll
    for (int tm = 0; tm < 4; ++tm) {
      v16bf af = frag_ld(sA[buf], wm + tm * 16, lane);
#pragma unroll
      for (int tn = 0; tn < 2; ++tn)
        acc[tm][tn] = __builtin_amdgcn_wmma_f32_16x16x32_bf16(
            false, af, false, bfr[tn], (short)0, acc[tm][tn], false, false);
    }
    __syncthreads();   // all waves done reading buf before it is overwritten
  }

  const int l15   = lane & 15;
  const int rhalf = (lane & 16) ? 8 : 0;
#pragma unroll
  for (int tm = 0; tm < 4; ++tm) {
#pragma unroll
    for (int tn = 0; tn < 2; ++tn) {
      int col  = bn + wn + tn * 16 + l15;
      int row0 = bm + wm + tm * 16 + rhalf;
      float bv = bias ? bias[col] : 0.f;
#pragma unroll
      for (int i = 0; i < 8; ++i) {
        float v = acc[tm][tn][i] + bv;
        if (relu) v = fmaxf(v, 0.f);
        size_t off = (size_t)(row0 + i) * N + col;
        if (C)   C[off]   = v;
        if (Cbf) Cbf[off] = f2bf(v);
      }
    }
  }
}

__global__ void k_f2bf(const float* __restrict__ x, unsigned short* __restrict__ y, size_t n) {
  size_t i = (size_t)blockIdx.x * 256 + threadIdx.x;
  if (i < n) y[i] = f2bf(x[i]);
}

__global__ void k_concat2(const float* __restrict__ a, const float* __restrict__ b,
                          float* __restrict__ o, int n1, int n2) {
  int i = blockIdx.x * 256 + threadIdx.x;
  if (i < n1) o[i] = a[i];
  else if (i < n1 + n2) o[i] = b[i - n1];
}

// actions [B,T,32] f32 -> [T,B,32] bf16
__global__ void k_build_a(const float* __restrict__ act, unsigned short* __restrict__ o) {
  int i = blockIdx.x * 256 + threadIdx.x;      // < T*B*32
  int c = i & 31;
  int r = i >> 5;          // t*B + b
  int b = r & 255;
  int t = r >> 8;
  o[i] = f2bf(act[((size_t)b * TT + t) * AK + c]);
}

// [a,o] concat: -> [T,B,1056] bf16
__global__ void k_build_xr(const float* __restrict__ act, const float* __restrict__ obs,
                           unsigned short* __restrict__ o, int n) {
  int i = blockIdx.x * 256 + threadIdx.x;
  if (i >= n) return;
  int c = i % 1056;
  int r = i / 1056;        // t*B + b
  int b = r & 255;
  int t = r >> 8;
  float v = (c < AK) ? act[((size_t)b * TT + t) * AK + c]
                     : obs[((size_t)b * TT + t) * DK + (c - AK)];
  o[i] = f2bf(v);
}

// Fused GRU gates + LayerNorm. One block per batch row (1024 cols, 256 thr x 4).
__global__ __launch_bounds__(256)
void k_gru_ln(const float* __restrict__ gi, const float* __restrict__ gh,
              const float* __restrict__ hprev, const float* __restrict__ g,
              const float* __restrict__ be, unsigned short* __restrict__ hd_bf) {
  __shared__ float rs[256], rq[256];
  int b = blockIdx.x, tid = threadIdx.x;
  const float* gib = gi + (size_t)b * G3;
  const float* ghb = gh + (size_t)b * G3;
  float v[4], s = 0.f, q = 0.f;
#pragma unroll
  for (int k = 0; k < 4; ++k) {
    int j = tid + k * 256;
    float r  = sigf(gib[j] + ghb[j]);
    float z  = sigf(gib[LK + j] + ghb[LK + j]);
    float nn = tanhf(gib[2 * LK + j] + r * ghb[2 * LK + j]);
    float val = (1.f - z) * nn + z * hprev[(size_t)b * LK + j];
    v[k] = val; s += val; q += val * val;
  }
  rs[tid] = s; rq[tid] = q; __syncthreads();
  for (int off = 128; off; off >>= 1) {
    if (tid < off) { rs[tid] += rs[tid + off]; rq[tid] += rq[tid + off]; }
    __syncthreads();
  }
  float mean = rs[0] * (1.f / 1024.f);
  float var  = rq[0] * (1.f / 1024.f) - mean * mean;
  float inv  = rsqrtf(var + 1e-5f);
#pragma unroll
  for (int k = 0; k < 4; ++k) {
    int j = tid + k * 256;
    float y = (v[k] - mean) * inv * g[j] + be[j];
    hd_bf[(size_t)b * LK + j] = f2bf(y);
  }
}

// mg = [B, 2048] (m | pre-softplus s). Writes outputs + f32/bf16 carries.
__global__ void k_sample(const float* __restrict__ mg, const float* __restrict__ eps,
                         int t, float* __restrict__ out_m, float* __restrict__ out_s,
                         float* __restrict__ carry_f, unsigned short* __restrict__ carry_bf,
                         float* __restrict__ out_sp, unsigned short* __restrict__ sp_bf) {
  int idx = blockIdx.x * 256 + threadIdx.x;    // < B*L
  int b = idx >> 10, j = idx & 1023;
  float pm = mg[(size_t)b * 2048 + j];
  float x  = mg[(size_t)b * 2048 + 1024 + j];
  float ps = (x > 20.f) ? x : log1pf(__expf(x));
  size_t po = ((size_t)b * TT + t) * LK + j;
  float sv = pm + ps * eps[po];
  out_m[po] = pm;
  out_s[po] = ps;
  carry_f[idx]  = sv;
  carry_bf[idx] = f2bf(sv);
  if (out_sp) { out_sp[po] = sv; sp_bf[po] = f2bf(sv); }
}

// rewards: one wave per row, dot(h[row, 0..K), w) + bias
__global__ __launch_bounds__(256)
void k_rewdot(const unsigned short* __restrict__ hbf, const float* __restrict__ w,
              const float* __restrict__ bias, float* __restrict__ out, int K) {
  int wid = threadIdx.x >> 5, lane = threadIdx.x & 31;
  int row = blockIdx.x * 8 + wid;
  const unsigned short* p = hbf + (size_t)row * K;
  float s = 0.f;
  for (int k = lane; k < K; k += 32) s += bf2f(p[k]) * w[k];
  for (int off = 16; off; off >>= 1) s += __shfl_xor(s, off, 32);
  if (lane == 0) out[row] = s + bias[0];
}

extern "C" void kernel_launch(void* const* d_in, const int* in_sizes, int n_in,
                              void* d_out, int out_size, void* d_ws, size_t ws_size,
                              hipStream_t stream) {
  const float* obs   = (const float*)d_in[0];
  const float* acts  = (const float*)d_in[1];
  const float* ep    = (const float*)d_in[2];
  const float* eq    = (const float*)d_in[3];
  const float* rW_ih = (const float*)d_in[4];
  const float* rW_hh = (const float*)d_in[5];
  const float* rb_ih = (const float*)d_in[6];
  const float* rb_hh = (const float*)d_in[7];
  const float* rln_g = (const float*)d_in[8];
  const float* rln_b = (const float*)d_in[9];
  const float* rWm   = (const float*)d_in[10];
  const float* rbm   = (const float*)d_in[11];
  const float* rWs   = (const float*)d_in[12];
  const float* rbs   = (const float*)d_in[13];
  const float* dW_ih = (const float*)d_in[14];
  const float* dW_hh = (const float*)d_in[15];
  const float* db_ih = (const float*)d_in[16];
  const float* db_hh = (const float*)d_in[17];
  const float* dln_g = (const float*)d_in[18];
  const float* dln_b = (const float*)d_in[19];
  const float* dWm   = (const float*)d_in[20];
  const float* dbm   = (const float*)d_in[21];
  const float* dWs   = (const float*)d_in[22];
  const float* dbs   = (const float*)d_in[23];
  const float* decW1 = (const float*)d_in[24];
  const float* decb1 = (const float*)d_in[25];
  const float* decW2 = (const float*)d_in[26];
  const float* decb2 = (const float*)d_in[27];
  const float* rewW1 = (const float*)d_in[28];
  const float* rewb1 = (const float*)d_in[29];
  const float* rewW2 = (const float*)d_in[30];
  const float* rewb2 = (const float*)d_in[31];

  float* out = (float*)d_out;
  const size_t NBT = (size_t)BT * LK;            // 16,777,216
  const size_t O_DEC = 0;
  const size_t O_REW = NBT;
  const size_t O_QM  = O_REW + BT;
  const size_t O_QS  = O_QM + NBT;
  const size_t O_PM  = O_QS + NBT;
  const size_t O_PS  = O_PM + NBT;
  const size_t O_HT  = O_PS + NBT;
  const size_t O_ST  = O_HT + (size_t)BB * LK;
  const size_t O_SP  = O_ST + (size_t)BB * LK;

  char* wp = (char*)d_ws;
  auto alloc = [&](size_t bytes) -> void* {
    void* r = (void*)wp;
    wp += (bytes + 255) & ~(size_t)255;
    return r;
  };
  // bf16 weights
  unsigned short* w_rih  = (unsigned short*)alloc((size_t)G3 * 1056 * 2);
  unsigned short* w_rhh  = (unsigned short*)alloc((size_t)G3 * LK * 2);
  unsigned short* w_dih  = (unsigned short*)alloc((size_t)G3 * AK * 2);
  unsigned short* w_dhh  = (unsigned short*)alloc((size_t)G3 * LK * 2);
  unsigned short* w_rms  = (unsigned short*)alloc((size_t)2 * LK * LK * 2); // [rWm;rWs]
  unsigned short* w_dms  = (unsigned short*)alloc((size_t)2 * LK * LK * 2); // [dWm;dWs]
  unsigned short* w_dec1 = (unsigned short*)alloc((size_t)HK * LK * 2);
  unsigned short* w_dec2 = (unsigned short*)alloc((size_t)DK * HK * 2);
  unsigned short* w_rew1 = (unsigned short*)alloc((size_t)HK * LK * 2);
  // bf16 activations
  unsigned short* a_tb   = (unsigned short*)alloc((size_t)TT * BB * AK * 2);
  unsigned short* xr_tb  = (unsigned short*)alloc((size_t)TT * BB * 1056 * 2);
  unsigned short* h_bf   = (unsigned short*)alloc((size_t)BB * LK * 2);
  unsigned short* s_bf   = (unsigned short*)alloc((size_t)BB * LK * 2);
  unsigned short* hd_bf  = (unsigned short*)alloc((size_t)BB * LK * 2);
  unsigned short* hr_bf  = (unsigned short*)alloc((size_t)BB * LK * 2);
  unsigned short* sp_bf  = (unsigned short*)alloc((size_t)BT * LK * 2);
  unsigned short* h1_bf  = (unsigned short*)alloc((size_t)BT * HK * 2);
  // f32 scratch
  float* gi_d = (float*)alloc((size_t)BB * G3 * 4);
  float* gi_r = (float*)alloc((size_t)BB * G3 * 4);
  float* gh_d = (float*)alloc((size_t)BB * G3 * 4);
  float* gh_r = (float*)alloc((size_t)BB * G3 * 4);
  float* pmps = (float*)alloc((size_t)BB * 2048 * 4);
  float* qmqs = (float*)alloc((size_t)BB * 2048 * 4);
  float* h_f  = (float*)alloc((size_t)BB * LK * 4);
  float* s_f  = (float*)alloc((size_t)BB * LK * 4);
  float* b_dms = (float*)alloc(2048 * 4);
  float* b_rms = (float*)alloc(2048 * 4);

  auto cvt = [&](const float* s, unsigned short* d, size_t n) {
    k_f2bf<<<dim3((unsigned)((n + 255) / 256)), dim3(256), 0, stream>>>(s, d, n);
  };
  auto gemm = [&](const unsigned short* A, const unsigned short* W, const float* bias,
                  float* C, unsigned short* Cbf, int M, int N, int K, int relu) {
    dim3 grid(N / 128, M / 128);
    k_gemm<<<grid, dim3(256), 0, stream>>>(A, W, bias, C, Cbf, M, N, K, relu);
  };

  // ---- weight prep ----
  cvt(rW_ih, w_rih, (size_t)G3 * 1056);
  cvt(rW_hh, w_rhh, (size_t)G3 * LK);
  cvt(dW_ih, w_dih, (size_t)G3 * AK);
  cvt(dW_hh, w_dhh, (size_t)G3 * LK);
  cvt(rWm, w_rms, (size_t)LK * LK);
  cvt(rWs, w_rms + (size_t)LK * LK, (size_t)LK * LK);
  cvt(dWm, w_dms, (size_t)LK * LK);
  cvt(dWs, w_dms + (size_t)LK * LK, (size_t)LK * LK);
  cvt(decW1, w_dec1, (size_t)HK * LK);
  cvt(decW2, w_dec2, (size_t)DK * HK);
  cvt(rewW1, w_rew1, (size_t)HK * LK);
  k_concat2<<<8, 256, 0, stream>>>(dbm, dbs, b_dms, LK, LK);
  k_concat2<<<8, 256, 0, stream>>>(rbm, rbs, b_rms, LK, LK);

  // ---- activation prep: [B,T,*] -> [T,B,*] bf16 ----
  k_build_a<<<(TT * BB * AK) / 256, 256, 0, stream>>>(acts, a_tb);
  {
    int n = TT * BB * 1056;
    k_build_xr<<<(n + 255) / 256, 256, 0, stream>>>(acts, obs, xr_tb, n);
  }

  // ---- zero carries ----
  hipMemsetAsync(h_f, 0, (size_t)BB * LK * 4, stream);
  hipMemsetAsync(s_f, 0, (size_t)BB * LK * 4, stream);
  hipMemsetAsync(h_bf, 0, (size_t)BB * LK * 2, stream);
  hipMemsetAsync(s_bf, 0, (size_t)BB * LK * 2, stream);

  // ---- sequential scan ----
  for (int t = 0; t < TT; ++t) {
    const unsigned short* a_t  = a_tb  + (size_t)t * BB * AK;
    const unsigned short* xr_t = xr_tb + (size_t)t * BB * 1056;

    // prior (dynamics) chain
    gemm(a_t, w_dih, db_ih, gi_d, nullptr, BB, G3, AK, 0);
    gemm(h_bf, w_dhh, db_hh, gh_d, nullptr, BB, G3, LK, 0);
    k_gru_ln<<<BB, 256, 0, stream>>>(gi_d, gh_d, h_f, dln_g, dln_b, hd_bf);
    gemm(hd_bf, w_dms, b_dms, pmps, nullptr, BB, 2048, LK, 0);
    k_sample<<<(BB * LK) / 256, 256, 0, stream>>>(
        pmps, ep, t, out + O_PM, out + O_PS, h_f, h_bf, nullptr, nullptr);

    // posterior (representation) chain
    gemm(xr_t, w_rih, rb_ih, gi_r, nullptr, BB, G3, 1056, 0);
    gemm(s_bf, w_rhh, rb_hh, gh_r, nullptr, BB, G3, LK, 0);
    k_gru_ln<<<BB, 256, 0, stream>>>(gi_r, gh_r, s_f, rln_g, rln_b, hr_bf);
    gemm(hr_bf, w_rms, b_rms, qmqs, nullptr, BB, 2048, LK, 0);
    k_sample<<<(BB * LK) / 256, 256, 0, stream>>>(
        qmqs, eq, t, out + O_QM, out + O_QS, s_f, s_bf, out + O_SP, sp_bf);
  }

  // ---- decoder / reward heads over all B*T tokens ----
  gemm(sp_bf, w_dec1, decb1, nullptr, h1_bf, BT, HK, LK, 1);          // relu -> bf16
  gemm(h1_bf, w_dec2, decb2, out + O_DEC, nullptr, BT, DK, HK, 0);    // decoded
  gemm(sp_bf, w_rew1, rewb1, nullptr, h1_bf, BT, HK, LK, 1);          // reuse h1_bf
  k_rewdot<<<BT / 8, 256, 0, stream>>>(h1_bf, rewW2, rewb2, out + O_REW, HK);

  // ---- final carries ----
  hipMemcpyAsync(out + O_HT, h_f, (size_t)BB * LK * 4, hipMemcpyDeviceToDevice, stream);
  hipMemcpyAsync(out + O_ST, s_f, (size_t)BB * LK * 4, hipMemcpyDeviceToDevice, stream);
}
